// Quasiseparable_68667937128994
// MI455X (gfx1250) — compile-verified, matrix operand order checked by reading, not run on hardware
//
#include <hip/hip_runtime.h>
#include <hip/hip_bf16.h>
#include <cmath>

// ---------------------------------------------------------------------------
// Bidirectional Mamba2-SSD (quasiseparable) for MI455X / gfx1250, wave32 WMMA.
// Precision: f16 inputs to v_wmma_f32_16x16x32_f16, f32 accumulation.
// ---------------------------------------------------------------------------

typedef __attribute__((ext_vector_type(16))) _Float16     v16h;
typedef __attribute__((ext_vector_type(8)))  float        v8f;
typedef __attribute__((ext_vector_type(4)))  unsigned int u32x4;
typedef __attribute__((ext_vector_type(4)))  float        f32x4;

#define SEQL   4096
#define CHUNK  256
#define NCH    16      // chunks
#define NH     16      // heads
#define HD     128     // head dim p
#define NS     256     // d_state n
#define DIN    2048
#define NB     2       // real batch
#define NBB    4       // batch*directions

static __device__ __forceinline__ v8f wmma16(v16h a, v16h b, v8f c) {
  return __builtin_amdgcn_wmma_f32_16x16x32_f16(false, a, false, b, (short)0, c, false, false);
}

// A/B fragment from LDS, operand stored [row(M or N)][K] row-major.
// lane&15 selects the row; lanes>=16 take K-base 8; element e: K = kb + (e<8 ? e : e+8).
static __device__ __forceinline__ v16h frag_lds(const _Float16* base, int row,
                                                int stride, int koff, int lane) {
  const int kb = (lane & 16) ? 8 : 0;
  const _Float16* p = base + row * stride + koff + kb;
  union { v16h v; u32x4 q[2]; } u;
  u.q[0] = *(const u32x4*)(p);       // K = kb .. kb+7
  u.q[1] = *(const u32x4*)(p + 16);  // K = kb+16 .. kb+23
  return u.v;
}

// A fragment with row data read directly from global f32 (row-major, contiguous K).
static __device__ __forceinline__ v16h fragA_g32(const float* rowbase, int koff, int lane) {
  const int kb = (lane & 16) ? 8 : 0;
  const float* p = rowbase + koff + kb;
  f32x4 a = *(const f32x4*)(p);
  f32x4 b = *(const f32x4*)(p + 4);
  f32x4 c = *(const f32x4*)(p + 16);
  f32x4 d = *(const f32x4*)(p + 20);
  v16h v;
  v[0]=(_Float16)a.x;  v[1]=(_Float16)a.y;  v[2]=(_Float16)a.z;  v[3]=(_Float16)a.w;
  v[4]=(_Float16)b.x;  v[5]=(_Float16)b.y;  v[6]=(_Float16)b.z;  v[7]=(_Float16)b.w;
  v[8]=(_Float16)c.x;  v[9]=(_Float16)c.y;  v[10]=(_Float16)c.z; v[11]=(_Float16)c.w;
  v[12]=(_Float16)d.x; v[13]=(_Float16)d.y; v[14]=(_Float16)d.z; v[15]=(_Float16)d.w;
  return v;
}

static __device__ __forceinline__ float softplusf(float x) {
  return fmaxf(x, 0.f) + log1pf(expf(-fabsf(x)));
}

// ---------------------------------------------------------------------------
// k_pre: softplus(dt), per-chunk inclusive cumsum cs, chunk decay.
// One thread per (bb, chunk j, head h): 1024 threads total.
// ---------------------------------------------------------------------------
__global__ __launch_bounds__(256) void k_pre(const float* __restrict__ dt,
                                             const float* __restrict__ A_log,
                                             float* __restrict__ w_arr,
                                             float* __restrict__ cs_arr,
                                             float* __restrict__ cd_arr) {
  int g  = blockIdx.x * 256 + threadIdx.x;       // 0..1023
  int bb = g >> 8, j = (g >> 4) & 15, h = g & 15;
  bool bw = bb >= 2;
  int  b_ = bb & 1;
  int  dcol = bw ? (NH + h) : h;
  float A = -expf(A_log[h]);
  int inst = (bb * NCH + j) * NH + h;
  float run = 0.f;
  for (int t = 0; t < CHUNK; ++t) {
    int tv = j * CHUNK + t;
    int st = bw ? (SEQL - 1 - tv) : tv;
    float w = softplusf(dt[((size_t)b_ * SEQL + st) * (2 * NH) + dcol]);
    run += w * A;
    w_arr[inst * CHUNK + t]  = w;
    cs_arr[inst * CHUNK + t] = run;
  }
  cd_arr[inst] = expf(run);
}

// ---------------------------------------------------------------------------
// k_states: per (bb,j,h), states[n,p] = sum_k B[k,n] * (exp(cs_last-cs_k)*w_k*x[k,p]).
// GEMM M=n=256, N=p=128, K=k=256 via WMMA f16. 1024 blocks x 256 threads.
// ---------------------------------------------------------------------------
__global__ __launch_bounds__(256) void k_states(const float* __restrict__ x,
                                                const float* __restrict__ BC,
                                                const float* __restrict__ w_arr,
                                                const float* __restrict__ cs_arr,
                                                _Float16* __restrict__ states) {
  const int tid = threadIdx.x, lane = tid & 31, wave = tid >> 5;
  const int bx = blockIdx.x;
  const int bb = bx >> 8, j = (bx >> 4) & 15, h = bx & 15;
  const bool bw = bb >= 2;
  const int  b_ = bb & 1;
  const int  inst = (bb * NCH + j) * NH + h;
  const int  half8 = (lane & 16) ? 8 : 0;

  __shared__ __align__(16) _Float16 BT[NS * 32];    // [n][k] 16KB
  __shared__ __align__(16) _Float16 XwT[HD * 32];   // [p][k]  8KB
  __shared__ float dsw[CHUNK];

  {
    float cl = cs_arr[inst * CHUNK + CHUNK - 1];
    float cq = cs_arr[inst * CHUNK + tid];
    dsw[tid] = expf(cl - cq) * w_arr[inst * CHUNK + tid];
  }

  v8f acc[2][8] = {};

  for (int ks = 0; ks < 8; ++ks) {
    __syncthreads();
    // stage Bt (32 k-rows x 256 n) transposed -> [n][k]
    for (int q = 0; q < 32; ++q) {
      int k = q, n = tid;
      int tv = j * CHUNK + ks * 32 + k;
      int st = bw ? (SEQL - 1 - tv) : tv;
      BT[n * 32 + k] = (_Float16)BC[((size_t)b_ * SEQL + st) * (2 * NS) + n];
    }
    // stage weighted X (32 k x 128 p) transposed -> [p][k]
    for (int q = 0; q < 16; ++q) {
      int e = tid + q * 256;
      int k = e >> 7, p = e & 127;
      int tv = j * CHUNK + ks * 32 + k;
      int st = bw ? (SEQL - 1 - tv) : tv;
      float v = x[((size_t)b_ * SEQL + st) * DIN + h * HD + p] * dsw[ks * 32 + k];
      XwT[p * 32 + k] = (_Float16)v;
    }
    __syncthreads();

    v16h a0 = frag_lds(BT, wave * 32 + (lane & 15), 32, 0, lane);
    v16h a1 = frag_lds(BT, wave * 32 + 16 + (lane & 15), 32, 0, lane);
    #pragma unroll
    for (int nt = 0; nt < 8; ++nt) {
      v16h bX = frag_lds(XwT, nt * 16 + (lane & 15), 32, 0, lane);
      acc[0][nt] = wmma16(a0, bX, acc[0][nt]);
      acc[1][nt] = wmma16(a1, bX, acc[1][nt]);
    }
  }

  const size_t sbase = (size_t)inst * NS * HD;
  #pragma unroll
  for (int mt = 0; mt < 2; ++mt)
    #pragma unroll
    for (int nt = 0; nt < 8; ++nt)
      #pragma unroll
      for (int r = 0; r < 8; ++r) {
        int n = wave * 32 + mt * 16 + r + half8;
        int p = nt * 16 + (lane & 15);
        states[sbase + (size_t)n * HD + p] = (_Float16)acc[mt][nt][r];
      }
}

// ---------------------------------------------------------------------------
// k_scan: exclusive scan of chunk states over the 16 chunks, per (bb, h).
// In-place: states[j] <- carry_before_j. 64 blocks x 1024 threads, f32 carry.
// ---------------------------------------------------------------------------
__global__ __launch_bounds__(1024) void k_scan(_Float16* __restrict__ states,
                                               const float* __restrict__ cd_arr) {
  const int tid = threadIdx.x;
  const int bb = blockIdx.x >> 4, h = blockIdx.x & 15;
  float c[32];
  #pragma unroll
  for (int q = 0; q < 32; ++q) c[q] = 0.f;

  for (int j = 0; j < NCH; ++j) {
    const int inst = (bb * NCH + j) * NH + h;
    const float cd = cd_arr[inst];
    _Float16* p = states + (size_t)inst * NS * HD + (size_t)tid * 32;
    #pragma unroll
    for (int q0 = 0; q0 < 32; q0 += 8) {
      union { u32x4 u; _Float16 hv[8]; } in, ot;
      in.u = *(const u32x4*)(p + q0);
      #pragma unroll
      for (int e = 0; e < 8; ++e) ot.hv[e] = (_Float16)c[q0 + e];
      *(u32x4*)(p + q0) = ot.u;
      #pragma unroll
      for (int e = 0; e < 8; ++e) c[q0 + e] = c[q0 + e] * cd + (float)in.hv[e];
    }
  }
}

// ---------------------------------------------------------------------------
// k_gate: out = x * repeat(x @ W_fcD^T + D, 128).  Full overwrite of d_out.
// Grid (SEQL, NB) x 256 threads.
// ---------------------------------------------------------------------------
__global__ __launch_bounds__(256) void k_gate(const float* __restrict__ x,
                                              const float* __restrict__ Dv,
                                              const float* __restrict__ W,
                                              float* __restrict__ out) {
  const int t = blockIdx.x, b_ = blockIdx.y, tid = threadIdx.x;
  __shared__ float xr[DIN];
  __shared__ float part[256];
  __shared__ float gate[NH];
  const float* xp = x + ((size_t)b_ * SEQL + t) * DIN;
  #pragma unroll
  for (int q = 0; q < 8; ++q) xr[tid + q * 256] = xp[tid + q * 256];
  __syncthreads();
  const int h = tid >> 4, sub = tid & 15;
  float s = 0.f;
  for (int m = 0; m < 128; ++m) {
    int d = sub + m * 16;
    s += xr[d] * W[(size_t)h * DIN + d];
  }
  part[tid] = s;
  __syncthreads();
  if (tid < NH) {
    float g = Dv[tid];
    #pragma unroll
    for (int s2 = 0; s2 < 16; ++s2) g += part[tid * 16 + s2];
    gate[tid] = g;
  }
  __syncthreads();
  float* op = out + ((size_t)b_ * SEQL + t) * DIN;
  #pragma unroll
  for (int q = 0; q < 8; ++q) {
    int d = tid + q * 256;
    op[d] = xr[d] * gate[d >> 7];
  }
}

// ---------------------------------------------------------------------------
// k_y: per (bb,j,h, 64-row block): y = (CB . decay . mask . w) @ X
//                                     + exp(cs_i) * (C @ prev)
// then scatter-add into d_out with the roll(+1)/flip mapping.
// Launched twice (fw: bb_base=0, bw: bb_base=2) so out RMW never races.
// Grid (64, 16, 2) x 256 threads (8 waves: 4 row-strips x 2 col-halves).
// ---------------------------------------------------------------------------
__global__ __launch_bounds__(256) void k_y(const float* __restrict__ x,
                                           const float* __restrict__ BC,
                                           const _Float16* __restrict__ states,
                                           const float* __restrict__ w_arr,
                                           const float* __restrict__ cs_arr,
                                           float* __restrict__ out,
                                           int bb_base) {
  const int tid = threadIdx.x, lane = tid & 31, wave = tid >> 5;
  const int wr = wave >> 1, wc = wave & 1;           // row-strip, col-half
  const int j = blockIdx.x >> 2, rb = blockIdx.x & 3;
  const int h = blockIdx.y;
  const int bb = bb_base + blockIdx.z;
  const bool bw = bb >= 2;
  const int  b_ = bb & 1;
  const int  R0 = rb * 64;
  const int  inst = (bb * NCH + j) * NH + h;
  const int  half8 = (lane & 16) ? 8 : 0;

  __shared__ __align__(16) _Float16 Bbuf[32 * NS];    // [k][n]   16KB
  __shared__ __align__(16) _Float16 XT[HD * 32];      // [p][k]    8KB
  __shared__ __align__(16) _Float16 PT[HD * 32];      // [p][n32]  8KB
  __shared__ __align__(16) _Float16 Sbuf[8 * 16 * 32];// per-wave  8KB
  __shared__ float eP[CHUNK], eN[CHUNK];

  {
    float cq = cs_arr[inst * CHUNK + tid];
    float wq = w_arr[inst * CHUNK + tid];
    eP[tid] = expf(cq);
    eN[tid] = expf(-cq) * wq;
  }
  __syncthreads();

  // per-lane C row pointer (row i is fixed per lane for all A fragments)
  const int iA = R0 + wr * 16 + (lane & 15);
  {
  }
  const int tvA = j * CHUNK + iA;
  const int stA = bw ? (SEQL - 1 - tvA) : tvA;
  const float* Crow = BC + ((size_t)b_ * SEQL + stA) * (2 * NS) + NS;

  // cache the whole 256-wide C row as 8 A-fragments (reused by stages D and A)
  v16h aC[8];
  #pragma unroll
  for (int ns = 0; ns < 8; ++ns) aC[ns] = fragA_g32(Crow, ns * 32, lane);

  float epi[8];
  #pragma unroll
  for (int r = 0; r < 8; ++r) epi[r] = eP[R0 + wr * 16 + r + half8];

  v8f Y[4] = {};

  // ---- stage D: Y = C @ prev  (scaled by exp(cs_i) afterwards) ----
  const size_t sbase = (size_t)inst * NS * HD;
  for (int nb = 0; nb < 8; ++nb) {
    __syncthreads();
    for (int q = 0; q < 16; ++q) {          // stage prev block transposed [p][n]
      int e = tid + q * 256;
      int nl = e >> 7, p = e & 127;
      PT[p * 32 + nl] = states[sbase + (size_t)(nb * 32 + nl) * HD + p];
    }
    __syncthreads();
    #pragma unroll
    for (int t = 0; t < 4; ++t) {
      v16h bP = frag_lds(PT, wc * 64 + t * 16 + (lane & 15), 32, 0, lane);
      Y[t] = wmma16(aC[nb], bP, Y[t]);
    }
  }
  #pragma unroll
  for (int t = 0; t < 4; ++t)
    #pragma unroll
    for (int r = 0; r < 8; ++r) Y[t][r] *= epi[r];

  // ---- stages A + C: causal (CB.decay.w) @ X, k-blocks of 32 ----
  const int kbmax = rb * 2 + 1;
  for (int kb = 0; kb <= kbmax; ++kb) {
    __syncthreads();
    for (int q = 0; q < 32; ++q) {          // stage B rows [k][n]
      int tvk = j * CHUNK + kb * 32 + q;
      int stk = bw ? (SEQL - 1 - tvk) : tvk;
      Bbuf[q * NS + tid] = (_Float16)BC[((size_t)b_ * SEQL + stk) * (2 * NS) + tid];
    }
    for (int q = 0; q < 16; ++q) {          // stage X transposed [p][k]
      int e = tid + q * 256;
      int k = e >> 7, p = e & 127;
      int tvk = j * CHUNK + kb * 32 + k;
      int stk = bw ? (SEQL - 1 - tvk) : tvk;
      XT[p * 32 + k] = (_Float16)x[((size_t)b_ * SEQL + stk) * DIN + h * HD + p];
    }
    __syncthreads();

    // CB tile: 16 rows x 32 k, K = n = 256
    v8f cb0 = {}, cb1 = {};
    #pragma unroll
    for (int ns = 0; ns < 8; ++ns) {
      v16h b0 = frag_lds(Bbuf, (lane & 15), NS, ns * 32, lane);
      v16h b1 = frag_lds(Bbuf, 16 + (lane & 15), NS, ns * 32, lane);
      cb0 = wmma16(aC[ns], b0, cb0);
      cb1 = wmma16(aC[ns], b1, cb1);
    }

    // decay + causal mask + dt weight, then park S in LDS (D-layout -> A-layout)
    const int k0 = kb * 32 + (lane & 15);
    const float en0 = eN[k0], en1 = eN[k0 + 16];
    _Float16* Sw = Sbuf + wave * 512;
    #pragma unroll
    for (int r = 0; r < 8; ++r) {
      int i = R0 + wr * 16 + r + half8;
      float s0 = (k0 <= i)      ? cb0[r] * epi[r] * en0 : 0.f;
      float s1 = (k0 + 16 <= i) ? cb1[r] * epi[r] * en1 : 0.f;
      Sw[(r + half8) * 32 + (lane & 15)]      = (_Float16)s0;
      Sw[(r + half8) * 32 + 16 + (lane & 15)] = (_Float16)s1;
    }

    v16h aS = frag_lds(Sw, (lane & 15), 32, 0, lane);
    #pragma unroll
    for (int t = 0; t < 4; ++t) {
      v16h bX = frag_lds(XT, wc * 64 + t * 16 + (lane & 15), 32, 0, lane);
      Y[t] = wmma16(aS, bX, Y[t]);
    }
  }

  // ---- epilogue: roll(+1) (fw) / flip+roll (bw) read-modify-write ----
  #pragma unroll
  for (int t = 0; t < 4; ++t)
    #pragma unroll
    for (int r = 0; r < 8; ++r) {
      int i = R0 + wr * 16 + r + half8;
      int tv = j * CHUNK + i;
      int ot = bw ? (SEQL - 2 - tv) : (tv + 1);
      if (ot < 0 || ot >= SEQL) continue;
      int col = h * HD + wc * 64 + t * 16 + (lane & 15);
      size_t idx = ((size_t)b_ * SEQL + ot) * DIN + col;
      out[idx] += Y[t][r];
    }
}

// ---------------------------------------------------------------------------
extern "C" void kernel_launch(void* const* d_in, const int* in_sizes, int n_in,
                              void* d_out, int out_size, void* d_ws, size_t ws_size,
                              hipStream_t stream) {
  const float* x     = (const float*)d_in[0];
  const float* BC    = (const float*)d_in[1];
  const float* dt    = (const float*)d_in[2];
  const float* A_log = (const float*)d_in[3];
  const float* Dv    = (const float*)d_in[4];
  const float* W     = (const float*)d_in[5];
  float* out = (float*)d_out;

  char* ws = (char*)d_ws;
  const size_t states_elems = (size_t)NBB * NCH * NH * NS * HD;   // 33.5M f16 = 64MB
  _Float16* states = (_Float16*)ws;
  float* w_arr  = (float*)(ws + states_elems * sizeof(_Float16));
  float* cs_arr = w_arr + (size_t)NBB * NCH * NH * CHUNK;
  float* cd_arr = cs_arr + (size_t)NBB * NCH * NH * CHUNK;

  k_pre<<<4, 256, 0, stream>>>(dt, A_log, w_arr, cs_arr, cd_arr);
  k_states<<<NBB * NCH * NH, 256, 0, stream>>>(x, BC, w_arr, cs_arr, states);
  k_scan<<<NBB * NH, 1024, 0, stream>>>(states, cd_arr);
  k_gate<<<dim3(SEQL, NB), 256, 0, stream>>>(x, Dv, W, out);
  k_y<<<dim3(NCH * 4, NH, 2), 256, 0, stream>>>(x, BC, states, w_arr, cs_arr, out, 0);
  k_y<<<dim3(NCH * 4, NH, 2), 256, 0, stream>>>(x, BC, states, w_arr, cs_arr, out, 2);
}
